// VectorQuantizer_87969520157318
// MI455X (gfx1250) — matrix operand order repaired and checked by reading, not compiled, not run on hardware
//
#include <hip/hip_runtime.h>
#include <hip/hip_bf16.h>

typedef __attribute__((ext_vector_type(16))) _Float16 v16h;
typedef __attribute__((ext_vector_type(8)))  _Float16 v8h;
typedef __attribute__((ext_vector_type(8)))  float    v8f;

#define NUM_CODES 1024
#define DIM       64
#define HW        4096          // H*W = 64*64
#define PER_B     262144        // DIM*HW floats per batch image
#define N_ELEMS   16777216      // 64*64*64*64
#define N_VEC     262144        // B*H*W rows
#define NUM_WG    2048          // N_VEC / 128 rows per workgroup

// dynamic LDS partition (bytes)
#define LDS_E_BYTES   (NUM_CODES * DIM * 2)          // 131072: f16 codebook
#define LDS_EN_OFF    LDS_E_BYTES                    // 4096:  -||e||^2/2
#define LDS_RED_OFF   (LDS_EN_OFF + NUM_CODES * 4)   // 32:    loss slots
#define LDS_TOTAL     (LDS_RED_OFF + 8 * 4)

// ---------------------------------------------------------------------------
// Kernel 0: codebook -> f16 (WMMA A operand) + per-code bias  -||e||^2 / 2
// ---------------------------------------------------------------------------
__global__ void vq_prep(const float* __restrict__ E,
                        _Float16* __restrict__ Ef16,
                        float* __restrict__ enh) {
  int code = blockIdx.x * blockDim.x + threadIdx.x;
  if (code >= NUM_CODES) return;
  const float* row = E + code * DIM;
  float s = 0.0f;
#pragma unroll
  for (int d = 0; d < DIM; ++d) {
    float v = row[d];
    s += v * v;
    Ef16[code * DIM + d] = (_Float16)v;
  }
  enh[code] = -0.5f * s;   // WMMA C-init: acc = z.e - ||e||^2/2  -> argmax
}

// ---------------------------------------------------------------------------
// Kernel 1: LDS-staged codebook, WMMA score sweep, fused argmax via C-init,
// gather + straight-through output + deterministic loss partial.
// Each wave owns 16 z-rows; block (8 waves) owns 128 consecutive rows.
// ---------------------------------------------------------------------------
__global__ __launch_bounds__(256)
void vq_main(const float* __restrict__ z,
             const float* __restrict__ E,
             const _Float16* __restrict__ Ef16,
             const float* __restrict__ enh,
             float* __restrict__ out_zq,
             float* __restrict__ out_idx,
             float* __restrict__ partials) {
  extern __shared__ char smem[];
  _Float16* ldsE  = (_Float16*)smem;
  float*    ldsEn = (float*)(smem + LDS_EN_OFF);
  float*    red   = (float*)(smem + LDS_RED_OFF);

  const int tid    = threadIdx.x;
  const int wave   = tid >> 5;
  const int lane   = tid & 31;
  const int lane16 = lane & 15;
  const bool hi    = lane >= 16;

  // ---- Stage f16 codebook (128KB) + bias (4KB) into LDS once per WG.
  {
    const uint4* src = (const uint4*)Ef16;
    uint4*       dst = (uint4*)ldsE;
#pragma unroll
    for (int i = 0; i < (LDS_E_BYTES / 16) / 256; ++i)
      dst[tid + 256 * i] = src[tid + 256 * i];
    ((float4*)ldsEn)[tid] = ((const float4*)enh)[tid];   // 1024 floats / 256 thr
  }

  // Row of z_flat this lane's WMMA column corresponds to
  const int nrow = blockIdx.x * 128 + wave * 16 + lane16;
  const int b    = nrow >> 12;      // 4096 rows per batch image
  const int np   = nrow & 4095;
  const float* zp = z + (size_t)b * PER_B + np;   // element d at zp[d*HW]

  // ---- Build B operands (z^T tile, 32 K x 16 rows, f16) once; reused for all
  // 64 code chunks. B layout: lanes 0-15 hold K {0..7,16..23}, lanes 16-31 K {8..15,24..31}.
  const int d0 = hi ? 8 : 0;
  v16h b_lo, b_hi;
#pragma unroll
  for (int j = 0; j < 16; ++j) {
    b_lo[j] = (_Float16)zp[(d0 + j + (j & 8)) * HW];        // K = 0..31 half
    b_hi[j] = (_Float16)zp[(d0 + 32 + j + (j & 8)) * HW];   // K = 32..63 half
  }

  __syncthreads();   // LDS codebook ready

  // A layout offsets: lanes<16 take K halves {0..7,16..23}, lanes>=16 {8..15,24..31}
  const int o0 = hi ? 8 : 0;

  float bestVal = -3.4028235e38f;
  int   bestIdx = 0;

  for (int c = 0; c < 64; ++c) {
    const _Float16* p = ldsE + (c * 16 + lane16) * DIM + o0;
    v8h x0 = *(const v8h*)(p);
    v8h x1 = *(const v8h*)(p + 16);
    v8h x2 = *(const v8h*)(p + 32);
    v8h x3 = *(const v8h*)(p + 48);
    v16h a_lo = __builtin_shufflevector(x0, x1, 0,1,2,3,4,5,6,7,8,9,10,11,12,13,14,15);
    v16h a_hi = __builtin_shufflevector(x2, x3, 0,1,2,3,4,5,6,7,8,9,10,11,12,13,14,15);

    // C-init with -||e||^2/2: after both WMMAs acc[r] = z.e - ||e||^2/2
    v8f acc = *(const v8f*)(ldsEn + c * 16 + o0);
    acc = __builtin_amdgcn_wmma_f32_16x16x32_f16(false, a_lo, false, b_lo,
                                                 (short)0, acc, false, false);
    acc = __builtin_amdgcn_wmma_f32_16x16x32_f16(false, a_hi, false, b_hi,
                                                 (short)0, acc, false, false);

    // D layout: lane holds column N=lane16; VGPR r -> code c*16 + o0 + r.
    // argmin(distance) == argmax(acc); strict > keeps first (lowest) index.
#pragma unroll
    for (int r = 0; r < 8; ++r) {
      float v = acc[r];
      if (v > bestVal) { bestVal = v; bestIdx = c * 16 + o0 + r; }
    }
  }

  // Merge the two half-lanes covering the same z-row (tie -> lower index).
  {
    float oval = __shfl_xor(bestVal, 16, 32);
    int   oidx = __shfl_xor(bestIdx, 16, 32);
    if (oval > bestVal || (oval == bestVal && oidx < bestIdx)) {
      bestVal = oval; bestIdx = oidx;
    }
  }

  if (!hi) out_idx[nrow] = (float)bestIdx;

  // ---- Gather winning code (f32), write straight-through output, loss part.
  const float* e  = E + (size_t)bestIdx * DIM;
  float*       op = out_zq + (size_t)b * PER_B + np;
  const int ds = hi ? 32 : 0;                     // lane pair splits the 64 dims
  float ls = 0.0f;
#pragma unroll
  for (int j = 0; j < 32; ++j) {
    int   d  = ds + j;
    float ev = e[d];
    float zv = zp[d * HW];
    op[d * HW] = zv + (ev - zv);                  // z + sg(z_q - z), ULP-faithful
    float df = zv - ev;
    ls += df * df;
  }

  // Deterministic reduction: wave shuffle tree -> per-wave slot -> serial sum.
#pragma unroll
  for (int off = 16; off > 0; off >>= 1) ls += __shfl_xor(ls, off, 32);
  if (lane == 0) red[wave] = ls;
  __syncthreads();
  if (tid == 0) {
    float s = 0.0f;
#pragma unroll
    for (int w = 0; w < 8; ++w) s += red[w];
    partials[blockIdx.x] = s;
  }
}

// ---------------------------------------------------------------------------
// Kernel 2: deterministic final loss reduction (single wave, fixed order)
// loss = (1 + 1 + 0.25) * mean((z - z_q)^2)
// ---------------------------------------------------------------------------
__global__ void vq_loss(const float* __restrict__ partials,
                        float* __restrict__ loss_out) {
  int lane = threadIdx.x;
  float s = 0.0f;
  for (int k = 0; k < NUM_WG / 32; ++k) s += partials[lane * (NUM_WG / 32) + k];
#pragma unroll
  for (int off = 16; off > 0; off >>= 1) s += __shfl_xor(s, off, 32);
  if (lane == 0) *loss_out = s * (2.25f / 16777216.0f);
}

// ---------------------------------------------------------------------------
extern "C" void kernel_launch(void* const* d_in, const int* in_sizes, int n_in,
                              void* d_out, int out_size, void* d_ws, size_t ws_size,
                              hipStream_t stream) {
  const float* z = (const float*)d_in[0];   // (64, 64, 64, 64) f32
  const float* E = (const float*)d_in[1];   // (1024, 64) f32

  char* ws = (char*)d_ws;
  _Float16* Ef16     = (_Float16*)(ws);               // 131072 B
  float*    enh      = (float*)(ws + 131072);         //   4096 B
  float*    partials = (float*)(ws + 135168);         //   8192 B

  float* out      = (float*)d_out;
  float* out_zq   = out;                    // 16777216 floats
  float* loss_out = out + N_ELEMS;          // 1 float
  float* out_idx  = out + N_ELEMS + 1;      // 262144 floats (indices as float)

  vq_prep<<<NUM_CODES / 256, 256, 0, stream>>>(E, Ef16, enh);
  vq_main<<<NUM_WG, 256, LDS_TOTAL, stream>>>(z, E, Ef16, enh,
                                              out_zq, out_idx, partials);
  vq_loss<<<1, 32, 0, stream>>>(partials, loss_out);
}
